// GNN_EdgeConv_3453153706715
// MI455X (gfx1250) — matrix operand order repaired and checked by reading, not compiled, not run on hardware
//
#include <hip/hip_runtime.h>

typedef __bf16 bf16;
typedef __attribute__((ext_vector_type(16))) __bf16 v16bf;
typedef __attribute__((ext_vector_type(8)))  __bf16 v8bf;
typedef __attribute__((ext_vector_type(8)))  float  v8f;

static constexpr int BB   = 16;    // batch
static constexpr int PP   = 1024;  // points
static constexpr int KNN  = 16;    // neighbors
static constexpr int HH   = 64;    // hidden
static constexpr int MID  = 96;    // edge-mlp mid width
static constexpr int DMAX = 192;

__device__ __forceinline__ v8f wmma_bf16(v16bf a, v16bf b, v8f c) {
  return __builtin_amdgcn_wmma_f32_16x16x32_bf16(false, a, false, b, (short)0, c,
                                                 false, false);
}

__device__ __forceinline__ v16bf cat8(v8bf lo, v8bf hi) {
  return __builtin_shufflevector(lo, hi, 0, 1, 2, 3, 4, 5, 6, 7, 8, 9, 10, 11,
                                 12, 13, 14, 15);
}

// issue a 16-byte async memory->LDS copy (ASYNCcnt-tracked)
__device__ __forceinline__ void async_b128(unsigned ldsOff, const void* gaddr) {
  asm volatile("global_load_async_to_lds_b128 %0, %1, off" ::"v"(ldsOff),
               "v"(gaddr)
               : "memory");
}
__device__ __forceinline__ void wait_async0() {
  asm volatile("s_wait_asynccnt 0x0" ::: "memory");
}
__device__ __forceinline__ unsigned lds_off(const void* p) {
  // generic pointers into the LDS aperture keep the LDS byte offset in [31:0]
  return (unsigned)(uintptr_t)p;
}

// ---------------------------------------------------------------- utilities
__global__ void k_zero_f32(float* __restrict__ p, int n) {
  int i = blockIdx.x * 256 + threadIdx.x;
  if (i < n) p[i] = 0.0f;
}

// pack a row-major f32 weight [Kdim x N] into WMMA-fragment-major bf16:
// dst[(((c*NT + nt)*32 + lane)*16) + t] = w[(c*32 + (lane>>4)*16 + t)*N + nt*16 + (lane&15)]
// so each lane's 16 B-fragment elements are contiguous (32 bytes).
__global__ void k_pack_w(const float* __restrict__ w, bf16* __restrict__ dst,
                         int Kdim, int N) {
  int idx = blockIdx.x * 256 + threadIdx.x;
  if (idx >= Kdim * N) return;
  int NT = N >> 4;
  int t = idx & 15;
  int l = (idx >> 4) & 31;
  int rest = idx >> 9;
  int nt = rest % NT;
  int c = rest / NT;
  int k = c * 32 + (l >> 4) * 16 + t;
  int n = nt * 16 + (l & 15);
  dst[idx] = (bf16)w[(size_t)k * N + n];
}

// --------------------------------------------------------------- input MLP
__global__ __launch_bounds__(128) void k_inmlp(
    const float* __restrict__ x, const float* __restrict__ dn,
    const float* __restrict__ w1, const float* __restrict__ b1,
    const float* __restrict__ w2, const float* __restrict__ b2,
    const float* __restrict__ w3, const float* __restrict__ b3,
    float* __restrict__ h0) {
  __shared__ float xin[5];
  __shared__ float l1[128];
  __shared__ float l2[128];
  int p = blockIdx.x, t = threadIdx.x;
  if (t < 5) xin[t] = x[(size_t)p * 5 + t] * dn[t];
  __syncthreads();
  {
    float a = b1[t];
#pragma unroll
    for (int k = 0; k < 5; ++k) a += xin[k] * w1[k * 128 + t];
    l1[t] = a > 0.f ? a : 0.f;
  }
  __syncthreads();
  {
    float a = b2[t];
    for (int k = 0; k < 128; ++k) a += l1[k] * w2[k * 128 + t];
    l2[t] = a > 0.f ? a : 0.f;
  }
  __syncthreads();
  if (t < 64) {
    float a = b3[t];
    for (int k = 0; k < 128; ++k) a += l2[k] * w3[k * 64 + t];
    h0[(size_t)p * 64 + t] = a > 0.f ? a : 0.f;
  }
}

// --------------------------------------------- L2-normalize + bf16 mirrors
__global__ __launch_bounds__(128) void k_norm(
    const float* __restrict__ h, bf16* __restrict__ hb, bf16* __restrict__ xn,
    int D) {
  int wave = threadIdx.x >> 5, lane = threadIdx.x & 31;
  int p = blockIdx.x * 4 + wave;
  const float* src = h + (size_t)p * D;
  float s = 0.f;
  for (int d = lane; d < D; d += 32) {
    float v = src[d];
    s += v * v;
  }
#pragma unroll
  for (int off = 16; off > 0; off >>= 1) s += __shfl_xor(s, off, 32);
  float rn = rsqrtf(s + 1e-12f);
  for (int d = lane; d < D; d += 32) {
    float v = src[d];
    hb[(size_t)p * D + d] = (bf16)v;
    xn[(size_t)p * D + d] = (bf16)(v * rn);
  }
}

// -------------------------------------------------- cosine-sim tile (WMMA)
__global__ __launch_bounds__(32) void k_sim_tile(
    const bf16* __restrict__ xn, float* __restrict__ sim, int D) {
  int rb = blockIdx.y * 16, cb = blockIdx.x * 16;
  int lane = threadIdx.x, half = lane >> 4, mrow = lane & 15;
  v8f acc = {};
  for (int kk = 0; kk < D; kk += 32) {
    const bf16* ap = xn + (size_t)(rb + mrow) * D + kk;
    v8bf lo = *(const v8bf*)(ap + half * 16);
    v8bf hi = *(const v8bf*)(ap + half * 16 + 8);
    v16bf a = cat8(lo, hi);
    const bf16* bp = xn + (size_t)(cb + mrow) * D + kk + half * 16;
    v16bf bf = *(const v16bf*)bp;
    acc = wmma_bf16(a, bf, acc);
  }
#pragma unroll
  for (int e = 0; e < 8; ++e) {
    int r = rb + e + half * 8;
    int c = cb + mrow;
    float v = acc[e];
    if (r == c) v -= 1e9f;
    sim[(size_t)r * PP + c] = v;
  }
}

// --------------------------------------------------------- top-k per row
// cache row in LDS once, then 16 argmax sweeps over LDS
__global__ __launch_bounds__(256) void k_topk(const float* __restrict__ sim,
                                              int* __restrict__ nbrOut) {
  __shared__ float row[PP];
  __shared__ float sv[256];
  __shared__ int si[256];
  int r0 = blockIdx.x;
  const float* src = sim + (size_t)r0 * PP;
  for (int c = threadIdx.x; c < PP; c += 256) row[c] = src[c];
  __syncthreads();
  for (int it = 0; it < KNN; ++it) {
    float best = -1e30f;
    int bi = PP;
    for (int c = threadIdx.x; c < PP; c += 256) {
      float v = row[c];
      if (v > best) { best = v; bi = c; }
    }
    sv[threadIdx.x] = best;
    si[threadIdx.x] = bi;
    __syncthreads();
    for (int off = 128; off > 0; off >>= 1) {
      if (threadIdx.x < off) {
        float ov = sv[threadIdx.x + off];
        int oi = si[threadIdx.x + off];
        if (ov > sv[threadIdx.x] ||
            (ov == sv[threadIdx.x] && oi < si[threadIdx.x])) {
          sv[threadIdx.x] = ov;
          si[threadIdx.x] = oi;
        }
      }
      __syncthreads();
    }
    if (threadIdx.x == 0) {
      nbrOut[r0 * KNN + it] = si[0];
      row[si[0]] = -1e30f;
    }
    __syncthreads();
  }
}

// ---------------------------------------------------------- mutual edges
__global__ void k_mutual(const int* __restrict__ nbr, int* __restrict__ mut) {
  int idx = blockIdx.x * 256 + threadIdx.x;
  if (idx >= BB * PP * KNN) return;
  int pnt = idx >> 4;
  int b = pnt >> 10, i = pnt & (PP - 1);
  int j = nbr[idx];
  const int* r = nbr + ((size_t)b * PP + j) * KNN;
  int f = 0;
#pragma unroll
  for (int t = 0; t < KNN; ++t) f |= (r[t] == i);
  mut[idx] = f;
}

// --------------------------------------------------- EdgeConv (WMMA core)
// one wave per center point; block = 4 waves.
//   GEMM1: e(16 x 2D) @ wa(2D x 96)   -> bias+BN+relu -> LDS (bf16)
//   GEMM2: h1(16 x 96) @ wb(96 x 64)  -> +bias -> relu -> aggregate
// neighbor features staged LDS via global_load_async_to_lds_b128
__global__ __launch_bounds__(128) void k_edgeconv(
    const bf16* __restrict__ hb, const int* __restrict__ nbr,
    const int* __restrict__ mut, const bf16* __restrict__ waP,
    const float* __restrict__ ba, const float* __restrict__ bnG,
    const float* __restrict__ bnB, const float* __restrict__ bnM,
    const float* __restrict__ bnV, const bf16* __restrict__ wbP,
    const float* __restrict__ bbias, float* __restrict__ outp, int D) {
  __shared__ alignas(32) bf16 s_xi[4][DMAX];
  __shared__ alignas(32) bf16 s_xj[4][KNN][DMAX];
  __shared__ alignas(32) bf16 s_h1[4][KNN][MID];
  __shared__ int s_nbr[4][KNN];
  __shared__ int s_mut[4][KNN];

  int wave = threadIdx.x >> 5, lane = threadIdx.x & 31;
  int half = lane >> 4, mrow = lane & 15;
  int center = blockIdx.x * 4 + wave;  // [0, B*P)
  int bIdx = center >> 10;

  if (lane < KNN) {
    s_nbr[wave][lane] = nbr[center * KNN + lane];
    s_mut[wave][lane] = mut[center * KNN + lane];
  }
  // ---- async stage center row (D*2 bytes in 16B chunks)
  {
    const char* g = (const char*)(hb + (size_t)center * D);
    const char* l = (const char*)&s_xi[wave][0];
    int nch = (D * 2) >> 4;
    for (int c = lane; c < nch; c += 32)
      async_b128(lds_off(l + c * 16), g + c * 16);
  }
  __syncthreads();  // s_nbr visible
  // ---- async stage 16 neighbor rows
  {
    int rowch = D >> 3;  // 16B chunks per row
    for (int c = lane; c < KNN * rowch; c += 32) {
      int r = c / rowch, cc = c - r * rowch;
      int j = s_nbr[wave][r];
      const char* g =
          (const char*)(hb + ((size_t)bIdx * PP + j) * D) + cc * 16;
      async_b128(lds_off((const char*)&s_xj[wave][r][0] + cc * 16), g);
    }
  }
  wait_async0();
  __syncthreads();

  for (int dir = 0; dir < 2; ++dir) {
    // ---- GEMM1: N = 96 in six 16-wide tiles, K = 2D in 32-chunks
    for (int nt = 0; nt < 6; ++nt) {
      v8f acc = {};
      for (int kk = 0; kk < 2 * D; kk += 32) {
        v16bf a;
#pragma unroll
        for (int t = 0; t < 16; ++t) {
          int k = kk + half * 16 + t;
          float val;
          if (dir == 0) {
            val = (k < D) ? (float)s_xi[wave][k]
                          : ((float)s_xj[wave][mrow][k - D] -
                             (float)s_xi[wave][k - D]);
          } else {
            val = (k < D) ? (float)s_xj[wave][mrow][k]
                          : ((float)s_xi[wave][k - D] -
                             (float)s_xj[wave][mrow][k - D]);
          }
          a[t] = (bf16)val;
        }
        v16bf bf =
            *(const v16bf*)(waP + ((size_t)(((kk >> 5) * 6 + nt) * 32 + lane)
                                   << 4));
        acc = wmma_bf16(a, bf, acc);
      }
      int c = nt * 16 + mrow;
      float scale = bnG[c] * rsqrtf(bnV[c] + 1e-5f);
      float bav = ba[c], mmv = bnM[c], bbv = bnB[c];
#pragma unroll
      for (int e = 0; e < 8; ++e) {
        float v = (acc[e] + bav - mmv) * scale + bbv;
        v = v > 0.f ? v : 0.f;
        s_h1[wave][e + half * 8][c] = (bf16)v;
      }
    }
    __syncthreads();

    // ---- GEMM2: (16 x 96) @ (96 x 64), four 16-wide N tiles
    for (int nt2 = 0; nt2 < 4; ++nt2) {
      v8f acc = {};
      for (int kk = 0; kk < MID; kk += 32) {
        v8bf lo = *(const v8bf*)&s_h1[wave][mrow][kk + half * 16];
        v8bf hi = *(const v8bf*)&s_h1[wave][mrow][kk + half * 16 + 8];
        v16bf a = cat8(lo, hi);
        v16bf bf =
            *(const v16bf*)(wbP + ((size_t)(((kk >> 5) * 4 + nt2) * 32 + lane)
                                   << 4));
        acc = wmma_bf16(a, bf, acc);
      }
      int c = nt2 * 16 + mrow;
      float bbv = bbias[c];
      if (dir == 0) {
        float s = 0.f;
#pragma unroll
        for (int e = 0; e < 8; ++e) {
          float v = acc[e] + bbv;
          s += v > 0.f ? v : 0.f;
        }
        s += __shfl_xor(s, 16, 32);
        if (half == 0) atomicAdd(&outp[(size_t)center * HH + c], s);
      } else {
#pragma unroll
        for (int e = 0; e < 8; ++e) {
          int r = e + half * 8;
          if (s_mut[wave][r]) continue;
          float v = acc[e] + bbv;
          v = v > 0.f ? v : 0.f;
          int j = s_nbr[wave][r];
          atomicAdd(&outp[((size_t)bIdx * PP + j) * HH + c], v);
        }
      }
    }
    __syncthreads();
  }
}

// ------------------------------------------------------------- concat
__global__ void k_concat(const float* __restrict__ ec,
                         const float* __restrict__ prev,
                         float* __restrict__ dst, int Dp, int n) {
  int idx = blockIdx.x * 256 + threadIdx.x;
  if (idx >= n) return;
  int Dt = HH + Dp;
  int pnt = idx / Dt, c = idx - pnt * Dt;
  dst[idx] = (c < HH) ? ec[(size_t)pnt * HH + c]
                      : prev[(size_t)pnt * Dp + (c - HH)];
}

// ------------------------------------------------------------- max pool
__global__ void k_pool(const float* __restrict__ cat3,
                       float* __restrict__ pooled) {
  int idx = blockIdx.x * 256 + threadIdx.x;
  if (idx >= BB * 256) return;
  int b = idx >> 8, c = idx & 255;
  float m = -1e30f;
  for (int p = 0; p < PP; ++p) {
    const float* q = cat3 + (((size_t)b * PP) + p) * 256 + c;
    __builtin_prefetch(q + 8 * 256, 0, 0);  // global_prefetch_b8
    float v = *q;
    m = v > m ? v : m;
  }
  pooled[idx] = m;
}

// ----------------------------------------------------------- output head
__global__ __launch_bounds__(128) void k_head(
    const float* __restrict__ pooled, const float* __restrict__ w1,
    const float* __restrict__ b1, const float* __restrict__ w2,
    const float* __restrict__ b2, const float* __restrict__ w3,
    const float* __restrict__ b3, float* __restrict__ outp) {
  __shared__ float pin[256];
  __shared__ float l1[128];
  __shared__ float l2[32];
  int b = blockIdx.x, t = threadIdx.x;
  pin[t] = pooled[b * 256 + t];
  pin[t + 128] = pooled[b * 256 + t + 128];
  __syncthreads();
  {
    float a = b1[t];
    for (int k = 0; k < 256; ++k) a += pin[k] * w1[k * 128 + t];
    l1[t] = a > 0.f ? a : 0.f;
  }
  __syncthreads();
  if (t < 32) {
    float a = b2[t];
    for (int k = 0; k < 128; ++k) a += l1[k] * w2[k * 32 + t];
    l2[t] = a > 0.f ? a : 0.f;
  }
  __syncthreads();
  if (t == 0) {
    float a = b3[0];
    for (int k = 0; k < 32; ++k) a += l2[k] * w3[k];
    outp[b] = a;
  }
}

// ==================================================================== host
extern "C" void kernel_launch(void* const* d_in, const int* in_sizes, int n_in,
                              void* d_out, int out_size, void* d_ws,
                              size_t ws_size, hipStream_t stream) {
  (void)in_sizes; (void)n_in; (void)out_size; (void)ws_size;
  const float* x = (const float*)d_in[0];
  const float* dn = (const float*)d_in[1];
  const float* iw1 = (const float*)d_in[2];
  const float* ib1 = (const float*)d_in[3];
  const float* iw2 = (const float*)d_in[4];
  const float* ib2 = (const float*)d_in[5];
  const float* iw3 = (const float*)d_in[6];
  const float* ib3 = (const float*)d_in[7];
  const float* cw[3][8];
  for (int l = 0; l < 3; ++l)
    for (int q = 0; q < 8; ++q) cw[l][q] = (const float*)d_in[8 + l * 8 + q];
  const float* ow1 = (const float*)d_in[32];
  const float* ob1 = (const float*)d_in[33];
  const float* ow2 = (const float*)d_in[34];
  const float* ob2 = (const float*)d_in[35];
  const float* ow3 = (const float*)d_in[36];
  const float* ob3 = (const float*)d_in[37];

  float* outv = (float*)d_out;  // [16] scores
  float* cat3 = outv + 16;      // [B,P,256] second output built in place

  char* w = (char*)d_ws;
  auto alloc = [&](size_t bytes) {
    char* p = w;
    w += (bytes + 255) & ~(size_t)255;
    return p;
  };
  const size_t NP = (size_t)BB * PP;
  float* h0 = (float*)alloc(NP * 64 * 4);
  float* cat1 = (float*)alloc(NP * 128 * 4);
  float* cat2 = (float*)alloc(NP * 192 * 4);
  float* ec = (float*)alloc(NP * 64 * 4);
  bf16* hb = (bf16*)alloc(NP * DMAX * 2);
  bf16* xn = (bf16*)alloc(NP * DMAX * 2);
  float* sim = (float*)alloc((size_t)PP * PP * 4);
  int* nbr = (int*)alloc(NP * KNN * 4);
  int* mut = (int*)alloc(NP * KNN * 4);
  bf16* waP = (bf16*)alloc((size_t)2 * DMAX * MID * 2);
  bf16* wbP = (bf16*)alloc((size_t)MID * HH * 2);
  float* pooled = (float*)alloc((size_t)BB * 256 * 4);

  auto cdiv = [](int a, int b) { return (a + b - 1) / b; };

  auto run_layer = [&](const float* hin, int D, const float* const* cp) {
    k_pack_w<<<cdiv(2 * D * MID, 256), 256, 0, stream>>>(cp[0], waP, 2 * D,
                                                         MID);
    k_pack_w<<<cdiv(MID * HH, 256), 256, 0, stream>>>(cp[6], wbP, MID, HH);
    k_norm<<<(int)(NP / 4), 128, 0, stream>>>(hin, hb, xn, D);
    for (int b = 0; b < BB; ++b) {
      dim3 g2(PP / 16, PP / 16);
      k_sim_tile<<<g2, 32, 0, stream>>>(xn + (size_t)b * PP * D, sim, D);
      k_topk<<<PP, 256, 0, stream>>>(sim, nbr + (size_t)b * PP * KNN);
    }
    k_mutual<<<cdiv(BB * PP * KNN, 256), 256, 0, stream>>>(nbr, mut);
    k_zero_f32<<<cdiv((int)(NP * 64), 256), 256, 0, stream>>>(ec,
                                                              (int)(NP * 64));
    k_edgeconv<<<(int)(NP / 4), 128, 0, stream>>>(hb, nbr, mut, waP, cp[1],
                                                  cp[2], cp[3], cp[4], cp[5],
                                                  wbP, cp[7], ec, D);
  };

  k_inmlp<<<(int)NP, 128, 0, stream>>>(x, dn, iw1, ib1, iw2, ib2, iw3, ib3,
                                       h0);

  run_layer(h0, 64, cw[0]);
  k_concat<<<cdiv((int)(NP * 128), 256), 256, 0, stream>>>(ec, h0, cat1, 64,
                                                           (int)(NP * 128));
  run_layer(cat1, 128, cw[1]);
  k_concat<<<cdiv((int)(NP * 192), 256), 256, 0, stream>>>(ec, cat1, cat2, 128,
                                                           (int)(NP * 192));
  run_layer(cat2, 192, cw[2]);
  k_concat<<<cdiv((int)(NP * 256), 256), 256, 0, stream>>>(ec, cat2, cat3, 192,
                                                           (int)(NP * 256));

  k_pool<<<cdiv(BB * 256, 256), 256, 0, stream>>>(cat3, pooled);
  k_head<<<BB, 128, 0, stream>>>(pooled, ow1, ob1, ow2, ob2, ow3, ob3, outv);
}
